// FrameStackEncoder_35450660062093
// MI455X (gfx1250) — matrix operand order repaired and checked by reading, not compiled, not run on hardware
//
#include <hip/hip_runtime.h>
#include <hip/hip_bf16.h>

// ---------------------------------------------------------------------------
// FrameStackEncoder for MI455X (gfx1250, wave32, WMMA + async-to-LDS).
// Pipeline (all on `stream`):
//   0) memset scratch (counts, vsum, BN stats, segmax)
//   1) one-time weight convert: f32 -> padded f16 (kills per-block cvt+2x traffic)
//   2) scatter: lin index, voxel counts (u32 atomics), voxel sums (f32 atomics)
//   3) feats:   8-ch point features + global BN stats (LDS ds_add_f32 -> global)
//   4) bnfin:   fold mean/var+gamma/beta into (scale, shift) per layer
//   5) GEMM x4: WMMA f32_16x16x32_f16, fused BN+ReLU on input, f16 activations,
//               double-buffered weight chunks via GLOBAL_LOAD_ASYNC_TO_LDS_B128
//               (compile-time probed; sync-copy fallback), per-channel stats;
//               last layer scatter-max into voxels
//   6) GEMM Wc: voxel 256->64 with mask+bias+ReLU
//   7) residual composition (B,192,H,W)
// ---------------------------------------------------------------------------

#define NPTS 524288
#define BB 2
#define FF 4
#define HH 64
#define WW 512
#define SS (BB * FF * HH * WW)   // 262144 voxels
#define EPSBN 1e-5f

typedef __attribute__((ext_vector_type(16))) _Float16 v16h;
typedef __attribute__((ext_vector_type(8)))  _Float16 v8h;
typedef __attribute__((ext_vector_type(8)))  float    v8f;
typedef int fse_v4i __attribute__((vector_size(16)));  // builtin's pointee type

// ===========================================================================
// Compile-time probe of the async-to-LDS builtin's accepted pointer flavors.
// ===========================================================================
#if __has_builtin(__builtin_amdgcn_global_load_async_to_lds_b128) && \
    __has_builtin(__builtin_amdgcn_s_wait_asynccnt)
#define FSE_ASYNC 1
#else
#define FSE_ASYNC 0
#endif

#if FSE_ASYNC
template <class...> using fse_void_t = void;

// variant GEN: plain generic v4i* arguments
template <typename T, typename = void>
struct fse_async_gen { static constexpr bool ok = false; };
template <typename T>
struct fse_async_gen<
    T, fse_void_t<decltype(__builtin_amdgcn_global_load_async_to_lds_b128(
           (fse_v4i*)(T*)nullptr, (fse_v4i*)(T*)nullptr, 0, 0))>> {
  static constexpr bool ok = true;
};

// variant AS: explicit address_space(1)/(3) qualified v4i* arguments
template <typename T, typename = void>
struct fse_async_as { static constexpr bool ok = false; };
template <typename T>
struct fse_async_as<
    T, fse_void_t<decltype(__builtin_amdgcn_global_load_async_to_lds_b128(
           (__attribute__((address_space(1))) fse_v4i*)(T*)nullptr,
           (__attribute__((address_space(3))) fse_v4i*)(T*)nullptr, 0, 0))>> {
  static constexpr bool ok = true;
};

constexpr bool FSE_AGEN = fse_async_gen<int>::ok;
constexpr bool FSE_AAS  = !FSE_AGEN && fse_async_as<int>::ok;
#else
constexpr bool FSE_AGEN = false;
constexpr bool FSE_AAS  = false;
#endif
constexpr bool FSE_USEASYNC = FSE_AGEN || FSE_AAS;

// ---- PROBE: fail device compilation loudly if the async path was NOT chosen.
#ifdef __HIP_DEVICE_COMPILE__
static_assert(FSE_USEASYNC,
              "FSE probe: async global->LDS path NOT selected on this toolchain");
#endif

// Issue one 16-byte async global->LDS transfer (variant chosen at compile time)
template <bool GEN, bool AS, typename T>
__device__ inline void fse_async_issue(const T* g, T* l) {
#if FSE_ASYNC
  if constexpr (GEN) {
    __builtin_amdgcn_global_load_async_to_lds_b128((fse_v4i*)g, (fse_v4i*)l, 0,
                                                   0);
  } else if constexpr (AS) {
    __builtin_amdgcn_global_load_async_to_lds_b128(
        (__attribute__((address_space(1))) fse_v4i*)(g),
        (__attribute__((address_space(3))) fse_v4i*)(l), 0, 0);
  }
#endif
}

__device__ inline void fse_wait_async() {
#if FSE_ASYNC
  __builtin_amdgcn_s_wait_asynccnt(0);
#endif
}

// ---- order-preserving float<->uint for atomicMax-based segment max ----------
__device__ inline unsigned f2ord(float f) {
  unsigned u = __float_as_uint(f);
  return (u & 0x80000000u) ? ~u : (u | 0x80000000u);
}
__device__ inline float ord2f(unsigned u) {
  unsigned b = (u & 0x80000000u) ? (u ^ 0x80000000u) : ~u;
  return __uint_as_float(b);
}

__device__ inline v8f zero8() {
  v8f z;
#pragma unroll
  for (int i = 0; i < 8; ++i) z[i] = 0.0f;
  return z;
}

// ---- WMMA fragment loaders (ISA 7.12.2 layouts, wave32) ---------------------
// A: 16x32 f16. lane&15 = row M, lane>>4 = K-half. VGPR v packs K pair:
//    k = k0 + 8*(lane>>4) + (v<4 ? 2v : 16+2(v-4))
__device__ inline v16h load_a_frag(const _Float16* sA, int lda, int rowbase,
                                   int k0, int lane) {
  const int m = lane & 15, hf = lane >> 4;
  const _Float16* p = sA + (rowbase + m) * lda + k0 + hf * 8;
  v16h a;
#pragma unroll
  for (int v = 0; v < 8; ++v) {
    const int kk = (v < 4) ? 2 * v : 16 + 2 * (v - 4);
    a[2 * v]     = p[kk];
    a[2 * v + 1] = p[kk + 1];
  }
  return a;
}

// B: 32x16 f16. lane&15 = column N, lane>>4 = K-half; element s has
//    k = 8*(lane>>4) + (s<8 ? s : s+8) (same K order as A).
__device__ inline v16h load_b_frag(const _Float16* sW, int ldn, int n0,
                                   int lane) {
  const int hf = lane >> 4;
  const _Float16* p = sW + n0 + (lane & 15);
  v16h b;
#pragma unroll
  for (int s = 0; s < 16; ++s) {
    const int kk = 8 * hf + (s < 8 ? s : s + 8);
    b[s] = p[kk * ldn];
  }
  return b;
}

// ---- weight-chunk staging: 32 x COUT f16 (64*COUT bytes), 16B granularity ---
template <int COUT, bool ASYNCV>
__device__ inline void stage_chunk(const _Float16* __restrict__ Whg,
                                   _Float16* sWbuf, int k0, int tid) {
  constexpr int NT16 = (32 * COUT) / 8;  // # of 16-byte transfers
  for (int t = tid; t < NT16; t += 128) {
    const _Float16* g = Whg + (size_t)k0 * COUT + t * 8;
    _Float16* l = sWbuf + t * 8;
    if constexpr (ASYNCV) {
      fse_async_issue<FSE_AGEN, FSE_AAS>(g, l);
    } else {
      *(v8h*)l = *(const v8h*)g;
    }
  }
}

// ---------------------------------------------------------------------------
// Kernel 0: one-time weight convert, f32 [CIN][COUT] -> f16 padded [CINP][COUT]
// ---------------------------------------------------------------------------
__global__ void fse_wcvt_kernel(const float* __restrict__ W,
                                _Float16* __restrict__ Wh, int CIN, int CINP,
                                int COUT) {
  const int i = blockIdx.x * blockDim.x + threadIdx.x;
  if (i >= CINP * COUT) return;
  const int k = i / COUT, n = i % COUT;
  Wh[i] = (k < CIN) ? (_Float16)W[(size_t)k * COUT + n] : (_Float16)0.0f;
}

// ---------------------------------------------------------------------------
// Kernel 1: voxel scatter (lin, counts, vsum)
// ---------------------------------------------------------------------------
__global__ void fse_scatter_kernel(const int* __restrict__ coors,
                                   const float* __restrict__ pts,
                                   unsigned* __restrict__ lin,
                                   unsigned* __restrict__ counts,
                                   float* __restrict__ vsum) {
  const int i = blockIdx.x * blockDim.x + threadIdx.x;
  if (i >= NPTS) return;
  const int b = coors[4 * i + 0], f = coors[4 * i + 1];
  const int h = coors[4 * i + 2], w = coors[4 * i + 3];
  const unsigned l = ((unsigned)(b * FF + f) * HH + h) * WW + w;
  lin[i] = l;
  atomicAdd(&counts[l], 1u);
#pragma unroll
  for (int c = 0; c < 4; ++c)
    atomicAdd(&vsum[(size_t)l * 4 + c], pts[4 * i + c]);
}

// ---------------------------------------------------------------------------
// Kernel 2: 8-channel features + global BN stats via LDS ds_add_f32
// ---------------------------------------------------------------------------
__global__ void fse_feats_kernel(const float* __restrict__ pts,
                                 const unsigned* __restrict__ lin,
                                 const unsigned* __restrict__ counts,
                                 const float* __restrict__ vsum,
                                 float* __restrict__ feats,
                                 float* __restrict__ ssum,
                                 float* __restrict__ ssq) {
  __shared__ float ls[16];
  const int tid = threadIdx.x;
  if (tid < 16) ls[tid] = 0.0f;
  __syncthreads();

  const int i = blockIdx.x * blockDim.x + tid;
  if (i < NPTS) {
    float p0 = pts[4 * i + 0], p1 = pts[4 * i + 1];
    float p2 = pts[4 * i + 2], p3 = pts[4 * i + 3];
    const unsigned l = lin[i];
    const float inv_c = 1.0f / (float)max(counts[l], 1u);
    const float dist = sqrtf(p0 * p0 + p1 * p1 + p2 * p2);
    float f8[8];
    f8[0] = p0; f8[1] = p1; f8[2] = p2; f8[3] = p3; f8[4] = dist;
    f8[5] = p0 - vsum[(size_t)l * 4 + 0] * inv_c;
    f8[6] = p1 - vsum[(size_t)l * 4 + 1] * inv_c;
    f8[7] = p2 - vsum[(size_t)l * 4 + 2] * inv_c;
#pragma unroll
    for (int c = 0; c < 8; ++c) {
      feats[(size_t)i * 8 + c] = f8[c];
      atomicAdd(&ls[c], f8[c]);
      atomicAdd(&ls[8 + c], f8[c] * f8[c]);
    }
  }
  __syncthreads();
  if (tid < 8) {
    atomicAdd(&ssum[tid], ls[tid]);
    atomicAdd(&ssq[tid], ls[8 + tid]);
  }
}

// ---------------------------------------------------------------------------
// Kernel 3: fold BN stats into (scale, shift)
// ---------------------------------------------------------------------------
__global__ void fse_bnfin_kernel(const float* __restrict__ sum,
                                 const float* __restrict__ sq,
                                 const float* __restrict__ gamma,
                                 const float* __restrict__ beta,
                                 float* __restrict__ scale,
                                 float* __restrict__ shift, int C, float invN) {
  const int c = blockIdx.x * blockDim.x + threadIdx.x;
  if (c >= C) return;
  const float m   = sum[c] * invN;
  const float var = sq[c] * invN - m * m;
  const float inv = rsqrtf(var + EPSBN);
  const float s   = gamma[c] * inv;
  scale[c] = s;
  shift[c] = beta[c] - m * s;
}

// ---------------------------------------------------------------------------
// Kernel 4: fused BN/ReLU + WMMA GEMM. 128 threads = 4 waves, 64 rows/block.
//   Weights: pre-converted padded f16, streamed as 32-K chunks through LDS
//   with async double-buffering (ASYNCcnt) when the builtin is usable.
//   MODE 0: in f32 feats, apply pre-BN (no relu), stats+store f16
//   MODE 1: in f16 y,     apply BN+relu,          stats+store f16
//   MODE 2: in f16 y3,    apply BN+relu, +bias,   scatter-max to segmax
//   MODE 3: in u32 segmax (masked decode), +bias+relu+mask, store f32 comp
// ---------------------------------------------------------------------------
template <int CIN, int CINP, int COUT, int MODE>
__global__ void fse_gemm_kernel(const void* __restrict__ xin,
                                const _Float16* __restrict__ Wh,  // [CINP][COUT]
                                const float* __restrict__ bias,
                                const float* __restrict__ scale,
                                const float* __restrict__ shift,
                                _Float16* __restrict__ yout,
                                float* __restrict__ stats_sum,
                                float* __restrict__ stats_sq,
                                const unsigned* __restrict__ lin,
                                const unsigned* __restrict__ counts,
                                unsigned* __restrict__ segmax,
                                float* __restrict__ comp) {
  extern __shared__ char smem[];
  _Float16* sA = (_Float16*)smem;            // 64 x CINP
  _Float16* sW0 = sA + 64 * CINP;            // 32 x COUT chunk, buffer 0
  _Float16* sW1 = sW0 + 32 * COUT;           // 32 x COUT chunk, buffer 1
  float* sSt   = (float*)(sW1 + 32 * COUT);  // 2 x COUT stats

  const int tid  = threadIdx.x;
  const int row0 = blockIdx.x * 64;

  // -------- stage input tile (BN-fold / ReLU / decode applied here) --------
  for (int i = tid; i < 64 * CINP; i += 128) {
    const int r = i / CINP, k = i % CINP;
    const size_t row = (size_t)(row0 + r);
    float v = 0.0f;
    if constexpr (MODE == 0) {
      if (k < CIN) v = ((const float*)xin)[row * CIN + k] * scale[k] + shift[k];
    } else if constexpr (MODE == 1 || MODE == 2) {
      float t = (float)((const _Float16*)xin)[row * CIN + k];
      t = t * scale[k] + shift[k];
      v = t > 0.0f ? t : 0.0f;
    } else {  // MODE 3: masked segment-max decode
      v = counts[row] ? ord2f(((const unsigned*)xin)[row * CIN + k]) : 0.0f;
    }
    sA[r * CINP + k] = (_Float16)v;
  }
  if constexpr (MODE <= 1) {
    for (int i = tid; i < 2 * COUT; i += 128) sSt[i] = 0.0f;
  }

  const int wave = tid >> 5, lane = tid & 31;
  const int rowbase = wave * 16;
  constexpr int NT = COUT / 16;
  v8f acc[NT];
#pragma unroll
  for (int j = 0; j < NT; ++j) acc[j] = zero8();

  if constexpr (FSE_USEASYNC) {
    // -------- async double-buffered K loop --------------------------------
    stage_chunk<COUT, true>(Wh, sW0, 0, tid);
    int buf = 0;
    for (int k0 = 0; k0 < CINP; k0 += 32) {
      fse_wait_async();  // this wave's async copies done
      __syncthreads();   // all waves' copies visible
      if (k0 + 32 < CINP)
        stage_chunk<COUT, true>(Wh, buf ? sW0 : sW1, k0 + 32, tid);
      const _Float16* sW = buf ? sW1 : sW0;
      const v16h a = load_a_frag(sA, CINP, rowbase, k0, lane);
#pragma unroll
      for (int j = 0; j < NT; ++j) {
        const v16h b = load_b_frag(sW, COUT, j * 16, lane);
        acc[j] = __builtin_amdgcn_wmma_f32_16x16x32_f16(
            false, a, false, b, (short)0, acc[j], false, false);
      }
      buf ^= 1;
    }
  } else {
    // -------- synchronous fallback ----------------------------------------
    for (int k0 = 0; k0 < CINP; k0 += 32) {
      __syncthreads();
      stage_chunk<COUT, false>(Wh, sW0, k0, tid);
      __syncthreads();
      const v16h a = load_a_frag(sA, CINP, rowbase, k0, lane);
#pragma unroll
      for (int j = 0; j < NT; ++j) {
        const v16h b = load_b_frag(sW0, COUT, j * 16, lane);
        acc[j] = __builtin_amdgcn_wmma_f32_16x16x32_f16(
            false, a, false, b, (short)0, acc[j], false, false);
      }
    }
  }

  // -------- epilogue ------------------------------------------------------
  const int hf = lane >> 4, ncol = lane & 15;
#pragma unroll
  for (int j = 0; j < NT; ++j) {
    const int n = j * 16 + ncol;
    float lsum = 0.0f, lsq = 0.0f;
#pragma unroll
    for (int v = 0; v < 8; ++v) {
      const size_t row = (size_t)row0 + rowbase + v + 8 * hf;
      float val = acc[j][v];
      if constexpr (MODE <= 1) {
        yout[row * COUT + n] = (_Float16)val;
        lsum += val;
        lsq  += val * val;
      } else if constexpr (MODE == 2) {
        val += bias[n];
        atomicMax(&segmax[(size_t)lin[row] * COUT + n], f2ord(val));
      } else {  // MODE 3
        val += bias[n];
        val = val > 0.0f ? val : 0.0f;
        if (!counts[row]) val = 0.0f;
        comp[row * COUT + n] = val;
      }
    }
    if constexpr (MODE <= 1) {
      atomicAdd(&sSt[n], lsum);          // ds_add_f32
      atomicAdd(&sSt[COUT + n], lsq);
    }
  }
  if constexpr (MODE <= 1) {
    __syncthreads();
    for (int i = tid; i < COUT; i += 128) {
      atomicAdd(&stats_sum[i], sSt[i]);
      atomicAdd(&stats_sq[i], sSt[COUT + i]);
    }
  }
}

// ---------------------------------------------------------------------------
// Kernel 5: residual frame stack -> (B, 192, H, W)
// ---------------------------------------------------------------------------
__global__ void fse_residual_kernel(const float* __restrict__ comp,
                                    float* __restrict__ out) {
  const int total = BB * 192 * HH * WW;
  const int idx = blockIdx.x * blockDim.x + threadIdx.x;
  if (idx >= total) return;
  const int w  = idx % WW;
  const int h  = (idx / WW) % HH;
  const int c  = (idx / (WW * HH)) % 192;
  const int b  = idx / (WW * HH * 192);
  const int f  = c / 64, cc = c % 64;
  const size_t last = (((size_t)b * FF + (FF - 1)) * HH + h) * WW + w;
  const size_t cur  = (((size_t)b * FF + f) * HH + h) * WW + w;
  out[idx] = comp[last * 64 + cc] - comp[cur * 64 + cc];
}

// ---------------------------------------------------------------------------
// Host launcher
// ---------------------------------------------------------------------------
extern "C" void kernel_launch(void* const* d_in, const int* in_sizes, int n_in,
                              void* d_out, int out_size, void* d_ws,
                              size_t ws_size, hipStream_t stream) {
  const float* pts       = (const float*)d_in[0];
  const int*   coors     = (const int*)d_in[1];
  const float* pre_gamma = (const float*)d_in[2];
  const float* pre_beta  = (const float*)d_in[3];
  const float* W1 = (const float*)d_in[4];
  const float* g1 = (const float*)d_in[5];
  const float* b1 = (const float*)d_in[6];
  const float* W2 = (const float*)d_in[7];
  const float* g2 = (const float*)d_in[8];
  const float* b2 = (const float*)d_in[9];
  const float* W3 = (const float*)d_in[10];
  const float* g3 = (const float*)d_in[11];
  const float* b3 = (const float*)d_in[12];
  const float* W4 = (const float*)d_in[13];
  const float* b4 = (const float*)d_in[14];
  const float* Wc = (const float*)d_in[15];
  const float* bc = (const float*)d_in[16];
  float* out = (float*)d_out;

  // ---- workspace layout (bytes) ----
  char* ws = (char*)d_ws;
  size_t off = 0;
  auto alloc = [&](size_t bytes) {
    char* p = ws + off;
    off += (bytes + 255) & ~(size_t)255;
    return p;
  };
  unsigned* lin    = (unsigned*)alloc((size_t)NPTS * 4);
  unsigned* counts = (unsigned*)alloc((size_t)SS * 4);
  float*    vsum   = (float*)   alloc((size_t)SS * 16);
  float*    stats  = (float*)   alloc(2048 * 4);
  _Float16* wh1    = (_Float16*)alloc(32 * 64 * 2);
  _Float16* wh2    = (_Float16*)alloc(64 * 128 * 2);
  _Float16* wh3    = (_Float16*)alloc(128 * 256 * 2);
  _Float16* wh4    = (_Float16*)alloc(256 * 256 * 2);
  _Float16* whc    = (_Float16*)alloc(256 * 64 * 2);
  float*    feats  = (float*)   alloc((size_t)NPTS * 8 * 4);
  _Float16* y1     = (_Float16*)alloc((size_t)NPTS * 64 * 2);
  _Float16* y2     = (_Float16*)alloc((size_t)NPTS * 128 * 2);
  _Float16* y3     = (_Float16*)alloc((size_t)NPTS * 256 * 2);
  unsigned* segmax = (unsigned*)alloc((size_t)SS * 256 * 4);
  float*    comp   = (float*)   alloc((size_t)SS * 64 * 4);

  // stats sub-layout (floats)
  float* sum_pre = stats + 0;    float* sq_pre = stats + 8;
  float* sum1 = stats + 16;      float* sq1 = stats + 80;
  float* sum2 = stats + 144;     float* sq2 = stats + 272;
  float* sum3 = stats + 400;     float* sq3 = stats + 656;
  float* sc_pre = stats + 912;   float* sh_pre = stats + 920;
  float* sc1 = stats + 928;      float* sh1 = stats + 992;
  float* sc2 = stats + 1056;     float* sh2 = stats + 1184;
  float* sc3 = stats + 1312;     float* sh3 = stats + 1568;

  const float invN = 1.0f / (float)NPTS;

  // 0) clear accumulators (segmax key 0 == minimum in ordered-uint space)
  (void)hipMemsetAsync(counts, 0, (size_t)SS * 4, stream);
  (void)hipMemsetAsync(vsum, 0, (size_t)SS * 16, stream);
  (void)hipMemsetAsync(stats, 0, 912 * 4, stream);
  (void)hipMemsetAsync(segmax, 0, (size_t)SS * 256 * 4, stream);

  // 1) one-time weight conversion to padded f16
  fse_wcvt_kernel<<<(32 * 64 + 255) / 256, 256, 0, stream>>>(W1, wh1, 8, 32, 64);
  fse_wcvt_kernel<<<(64 * 128 + 255) / 256, 256, 0, stream>>>(W2, wh2, 64, 64, 128);
  fse_wcvt_kernel<<<(128 * 256 + 255) / 256, 256, 0, stream>>>(W3, wh3, 128, 128, 256);
  fse_wcvt_kernel<<<(256 * 256 + 255) / 256, 256, 0, stream>>>(W4, wh4, 256, 256, 256);
  fse_wcvt_kernel<<<(256 * 64 + 255) / 256, 256, 0, stream>>>(Wc, whc, 256, 256, 64);

  // 2) voxel scatter
  fse_scatter_kernel<<<NPTS / 256, 256, 0, stream>>>(coors, pts, lin, counts,
                                                     vsum);
  // 3) features + pre-BN stats
  fse_feats_kernel<<<NPTS / 256, 256, 0, stream>>>(pts, lin, counts, vsum,
                                                   feats, sum_pre, sq_pre);
  fse_bnfin_kernel<<<1, 256, 0, stream>>>(sum_pre, sq_pre, pre_gamma, pre_beta,
                                          sc_pre, sh_pre, 8, invN);

  // 4) MLP via WMMA (shared mem = A tile + 2 weight chunks + stats)
  auto smem = [](int cinp, int cout, bool st) {
    return (size_t)(64 * cinp * 2 + 2 * 32 * cout * 2 +
                    (st ? 2 * cout * 4 : 0));
  };
  fse_gemm_kernel<8, 32, 64, 0><<<NPTS / 64, 128, smem(32, 64, true), stream>>>(
      feats, wh1, nullptr, sc_pre, sh_pre, y1, sum1, sq1, nullptr, nullptr,
      nullptr, nullptr);
  fse_bnfin_kernel<<<1, 256, 0, stream>>>(sum1, sq1, g1, b1, sc1, sh1, 64, invN);

  fse_gemm_kernel<64, 64, 128, 1>
      <<<NPTS / 64, 128, smem(64, 128, true), stream>>>(
          y1, wh2, nullptr, sc1, sh1, y2, sum2, sq2, nullptr, nullptr, nullptr,
          nullptr);
  fse_bnfin_kernel<<<1, 256, 0, stream>>>(sum2, sq2, g2, b2, sc2, sh2, 128,
                                          invN);

  fse_gemm_kernel<128, 128, 256, 1>
      <<<NPTS / 64, 128, smem(128, 256, true), stream>>>(
          y2, wh3, nullptr, sc2, sh2, y3, sum3, sq3, nullptr, nullptr, nullptr,
          nullptr);
  fse_bnfin_kernel<<<1, 256, 0, stream>>>(sum3, sq3, g3, b3, sc3, sh3, 256,
                                          invN);

  // 5) layer 4 + fused scatter-max into voxels
  fse_gemm_kernel<256, 256, 256, 2>
      <<<NPTS / 64, 128, smem(256, 256, false), stream>>>(
          y3, wh4, b4, sc3, sh3, nullptr, nullptr, nullptr, lin, counts, segmax,
          nullptr);

  // 6) voxel compression GEMM (masked decode of segmax)
  fse_gemm_kernel<256, 256, 64, 3>
      <<<SS / 64, 128, smem(256, 64, false), stream>>>(
          segmax, whc, bc, nullptr, nullptr, nullptr, nullptr, nullptr, nullptr,
          counts, nullptr, comp);

  // 7) residual frame stack
  const int total = BB * 192 * HH * WW;
  fse_residual_kernel<<<(total + 255) / 256, 256, 0, stream>>>(comp, out);
}
